// BaselineGCN_82188494176331
// MI455X (gfx1250) — matrix operand order repaired and checked by reading, not compiled, not run on hardware
//
#include <hip/hip_runtime.h>
#include <hip/hip_bf16.h>
#include <stdint.h>

// ---------- types ----------
typedef __attribute__((ext_vector_type(16))) __bf16 v16bf;
typedef __attribute__((ext_vector_type(8)))  __bf16 v8bf;
typedef __attribute__((ext_vector_type(8)))  float  v8f;

#define GCN_N    50000
#define GCN_E    800000
#define GCN_DIN  256
#define GCN_DH   256
#define GCN_DOUT 64

__device__ __forceinline__ uint16_t f2bf_bits(float f) {
  union { float f; uint32_t u; } v; v.f = f;
  return (uint16_t)((v.u + 0x7FFFu + ((v.u >> 16) & 1u)) >> 16);  // RNE
}

__device__ __forceinline__ void atomic_add_f32(float* p, float v) {
  __hip_atomic_fetch_add(p, v, __ATOMIC_RELAXED, __HIP_MEMORY_SCOPE_AGENT);
}

// ---------- degree / normalization ----------
__global__ void init_deg_k(float* deg, int n) {
  int i = blockIdx.x * 256 + threadIdx.x;
  if (i < n) deg[i] = 1.0f;  // self loop contributes 1
}

__global__ void edge_deg_k(float* deg, const int* __restrict__ dst, int e) {
  int i = blockIdx.x * 256 + threadIdx.x;
  if (i < e) atomic_add_f32(&deg[dst[i]], 1.0f);
}

__global__ void rsqrt_k(const float* __restrict__ deg, float* __restrict__ dis, int n) {
  int i = blockIdx.x * 256 + threadIdx.x;
  if (i < n) dis[i] = rsqrtf(deg[i]);  // deg >= 1 always
}

// ---------- weight transpose + bf16 convert:  Wt[n*K + k] = W[k*Nout + n] ----------
__global__ void wt_bf16_k(const float* __restrict__ W, uint16_t* __restrict__ Wt,
                          int K, int Nout) {
  int idx = blockIdx.x * 256 + threadIdx.x;
  if (idx >= K * Nout) return;
  int n = idx / K;
  int k = idx - n * K;
  Wt[idx] = f2bf_bits(W[(size_t)k * Nout + n]);
}

// ---------- activation fp32 -> bf16 (optionally fused ReLU), 4 elems/thread ----------
__global__ void cvt_bf16_k(const float* __restrict__ in, uint16_t* __restrict__ out,
                           int n4, int relu) {
  int i = blockIdx.x * 256 + threadIdx.x;
  if (i >= n4) return;
  float4 v = ((const float4*)in)[i];
  if (relu) {
    v.x = fmaxf(v.x, 0.0f); v.y = fmaxf(v.y, 0.0f);
    v.z = fmaxf(v.z, 0.0f); v.w = fmaxf(v.w, 0.0f);
  }
  ushort4 o;
  o.x = f2bf_bits(v.x); o.y = f2bf_bits(v.y);
  o.z = f2bf_bits(v.z); o.w = f2bf_bits(v.w);
  ((ushort4*)out)[i] = o;
}

// ---------- WMMA GEMM: H[M,Nout] = A[M,K](bf16) * Wt^T  (Wt is [Nout,K] bf16) ----------
// One wave computes a 16x64 tile: the A fragment is loaded once per k-step and
// reused across 4 B fragments (4 accumulators) -> 32 v_wmma per wave, 4x less
// A traffic. K == 256 fixed.
// Fragment layouts used here (contiguous k order, same for A and B):
//   A:   lane r / r+16 holds row m0+r; elems 0..15 = K(half*16 + 0..15)
//   B:   lane r holds column n+r; elems 0..15 = K(half*16 + 0..15) (contiguous in Wt)
//   C/D: vgpr j -> row m0 + half*8 + j, col n + r
template <int NWAVES>
__global__ void gemm_bf16_wmma_k(const uint16_t* __restrict__ A,
                                 const uint16_t* __restrict__ Wt,
                                 float* __restrict__ H, int K, int Nout) {
  const int lane = threadIdx.x & 31;
  const int wave = threadIdx.x >> 5;
  const int r    = lane & 15;
  const int half = lane >> 4;
  const int m0   = blockIdx.x * 16;
  const int n0   = (blockIdx.y * NWAVES + wave) * 64;  // 4 column tiles per wave

  const __bf16* aptr = (const __bf16*)(A  + (size_t)(m0 + r) * K) + half * 16;
  const __bf16* bptr = (const __bf16*)(Wt + (size_t)(n0 + r) * K) + half * 16;

  v8f c0 = {}, c1 = {}, c2 = {}, c3 = {};
#pragma unroll
  for (int kk = 0; kk < 8; ++kk) {  // K = 256
    v8bf  alo = *(const v8bf*)(aptr + kk * 32);
    v8bf  ahi = *(const v8bf*)(aptr + kk * 32 + 8);
    v16bf a;
#pragma unroll
    for (int i = 0; i < 8; ++i) { a[i] = alo[i]; a[8 + i] = ahi[i]; }
    // B fragments for column tiles n0, n0+16, n0+32, n0+48: constant byte
    // offsets q*16*K*2 = q*8192 (fits 24-bit instruction offset).
    v16bf b0 = *(const v16bf*)(bptr + 0 * 16 * 256 + kk * 32);
    v16bf b1 = *(const v16bf*)(bptr + 1 * 16 * 256 + kk * 32);
    v16bf b2 = *(const v16bf*)(bptr + 2 * 16 * 256 + kk * 32);
    v16bf b3 = *(const v16bf*)(bptr + 3 * 16 * 256 + kk * 32);
    c0 = __builtin_amdgcn_wmma_f32_16x16x32_bf16(false, a, false, b0, (short)0, c0, false, false);
    c1 = __builtin_amdgcn_wmma_f32_16x16x32_bf16(false, a, false, b1, (short)0, c1, false, false);
    c2 = __builtin_amdgcn_wmma_f32_16x16x32_bf16(false, a, false, b2, (short)0, c2, false, false);
    c3 = __builtin_amdgcn_wmma_f32_16x16x32_bf16(false, a, false, b3, (short)0, c3, false, false);
  }

  float* outp = H + (size_t)(m0 + half * 8) * Nout + n0 + r;
#pragma unroll
  for (int j = 0; j < 8; ++j) {
    outp[(size_t)j * Nout +  0] = c0[j];
    outp[(size_t)j * Nout + 16] = c1[j];
    outp[(size_t)j * Nout + 32] = c2[j];
    outp[(size_t)j * Nout + 48] = c3[j];
  }
}

// ---------- propagation: self-loop + bias init, then edge scatter-add ----------
__global__ void prop_init_k(const float* __restrict__ H, const float* __restrict__ dis,
                            const float* __restrict__ bias, float* __restrict__ P,
                            int n, int D) {
  int t  = blockIdx.x * 256 + threadIdx.x;
  int pe = D >> 2;
  if (t >= n * pe) return;
  int i = t / pe;
  int f = (t - i * pe) << 2;
  float s = dis[i]; s = s * s;
  float4 h  = *(const float4*)(H + (size_t)i * D + f);
  float4 bb = *(const float4*)(bias + f);
  float4 o;
  o.x = h.x * s + bb.x; o.y = h.y * s + bb.y;
  o.z = h.z * s + bb.z; o.w = h.w * s + bb.w;
  *(float4*)(P + (size_t)i * D + f) = o;
}

__global__ void prop_edge_k(const float* __restrict__ H, const float* __restrict__ dis,
                            const int* __restrict__ src, const int* __restrict__ dst,
                            float* __restrict__ P, int e, int D) {
  int t  = blockIdx.x * 256 + threadIdx.x;
  int pe = D >> 2;
  if (t >= e * pe) return;  // max 51.2M, fits int
  int ed = t / pe;
  int f  = (t - ed * pe) << 2;
  int s  = src[ed];
  int d  = dst[ed];
  float w = dis[s] * dis[d];
  float4 h = *(const float4*)(H + (size_t)s * D + f);
  float* o = P + (size_t)d * D + f;
  atomic_add_f32(o + 0, h.x * w);
  atomic_add_f32(o + 1, h.y * w);
  atomic_add_f32(o + 2, h.z * w);
  atomic_add_f32(o + 3, h.w * w);
}

// ---------- host ----------
extern "C" void kernel_launch(void* const* d_in, const int* in_sizes, int n_in,
                              void* d_out, int out_size, void* d_ws, size_t ws_size,
                              hipStream_t stream) {
  const float* x  = (const float*)d_in[0];
  const int*   ei = (const int*)  d_in[1];  // [2, E]: row0 = src, row1 = dst
  const float* W1 = (const float*)d_in[2];
  const float* b1 = (const float*)d_in[3];
  const float* W2 = (const float*)d_in[4];
  const float* b2 = (const float*)d_in[5];
  const float* W3 = (const float*)d_in[6];
  const float* b3 = (const float*)d_in[7];
  float* out = (float*)d_out;

  const int N = GCN_N, E = GCN_E;
  const int* src = ei;
  const int* dst = ei + E;

  char* ws = (char*)d_ws;
  size_t off = 0;
  auto alloc = [&](size_t bytes) -> void* {
    void* p = ws + off;
    off = (off + bytes + 255) & ~(size_t)255;
    return p;
  };
  float*    deg = (float*)   alloc((size_t)N * 4);
  float*    dis = (float*)   alloc((size_t)N * 4);
  uint16_t* Wt1 = (uint16_t*)alloc((size_t)256 * 256 * 2);
  uint16_t* Wt2 = (uint16_t*)alloc((size_t)256 * 256 * 2);
  uint16_t* Wt3 = (uint16_t*)alloc((size_t)256 * 64 * 2);
  uint16_t* Ab  = (uint16_t*)alloc((size_t)N * 256 * 2);
  float*    Hf  = (float*)   alloc((size_t)N * 256 * 4);
  float*    P   = (float*)   alloc((size_t)N * 256 * 4);

  const dim3 blk(256);
  const int nElems = N * 256;        // 12.8M
  const int cvtBlocks  = (nElems / 4 + 255) / 256;
  const int propBlocks = (N * 64 + 255) / 256;       // N * D/4, D=256
  const int edgeBlocks = (E * 64 + 255) / 256;       // E * D/4, D=256

  // normalization
  init_deg_k<<<(N + 255) / 256, blk, 0, stream>>>(deg, N);
  edge_deg_k<<<(E + 255) / 256, blk, 0, stream>>>(deg, dst, E);
  rsqrt_k  <<<(N + 255) / 256, blk, 0, stream>>>(deg, dis, N);

  // weights -> bf16, column-major
  wt_bf16_k<<<(256 * 256 + 255) / 256, blk, 0, stream>>>(W1, Wt1, 256, 256);
  wt_bf16_k<<<(256 * 256 + 255) / 256, blk, 0, stream>>>(W2, Wt2, 256, 256);
  wt_bf16_k<<<(256 * 64  + 255) / 256, blk, 0, stream>>>(W3, Wt3, 256, 64);

  // ---- layer 1 ----
  cvt_bf16_k<<<cvtBlocks, blk, 0, stream>>>(x, Ab, nElems / 4, 0);
  gemm_bf16_wmma_k<4><<<dim3(N / 16, 1), dim3(128), 0, stream>>>(Ab, Wt1, Hf, 256, 256);
  prop_init_k<<<propBlocks, blk, 0, stream>>>(Hf, dis, b1, P, N, 256);
  prop_edge_k<<<edgeBlocks, blk, 0, stream>>>(Hf, dis, src, dst, P, E, 256);

  // ---- layer 2 ---- (ReLU fused into bf16 conversion)
  cvt_bf16_k<<<cvtBlocks, blk, 0, stream>>>(P, Ab, nElems / 4, 1);
  gemm_bf16_wmma_k<4><<<dim3(N / 16, 1), dim3(128), 0, stream>>>(Ab, Wt2, Hf, 256, 256);
  prop_init_k<<<propBlocks, blk, 0, stream>>>(Hf, dis, b2, P, N, 256);
  prop_edge_k<<<edgeBlocks, blk, 0, stream>>>(Hf, dis, src, dst, P, E, 256);

  // ---- layer 3 ---- (no ReLU on output; writes d_out; single wave covers 64 cols)
  cvt_bf16_k<<<cvtBlocks, blk, 0, stream>>>(P, Ab, nElems / 4, 1);
  gemm_bf16_wmma_k<1><<<dim3(N / 16, 1), dim3(32), 0, stream>>>(Ab, Wt3, Hf, 256, 64);
  prop_init_k<<<(N * 16 + 255) / 256, blk, 0, stream>>>(Hf, dis, b3, out, N, 64);
  prop_edge_k<<<(E * 16 + 255) / 256, blk, 0, stream>>>(Hf, dis, src, dst, out, E, 64);
}
